// ConvShapeletFilterClassWise_74912819576991
// MI455X (gfx1250) — matrix operand order repaired and checked by reading, not compiled, not run on hardware
//
#include <hip/hip_runtime.h>

typedef __attribute__((ext_vector_type(16))) _Float16 v16h;
typedef __attribute__((ext_vector_type(8)))  float    v8f;

#define T_LEN   4096
#define L_LEN   128
#define NSH     128
#define NWIN    3969      // 4096 - 128 + 1
#define NBLK    249       // ceil(3969/16)
#define REPS    4         // shifted LDS replicas -> 8B-aligned A loads
#define RSTRIDE 4112      // 4096 + pad, multiple of 8
#define EPS_F   1e-6f
#define NEG_BIG -3.0e38f

union V16HU { v16h v; uint4 u4[2]; };

// ---------------------------------------------------------------------------
// Kernel 1: z-normalize shapelets and pack into WMMA B-operand layout.
// B operand (16x16x32 f16, 32x16 B): lane l (0-15): N=l, elem j -> K=j (j<8), K=j+8 (j>=8);
// lanes 16-31: N=l-16, elem j -> K=8+j (j<8), K=16+j (j>=8).  spack[(ntile*4+kc)*32 + lane] = v16h.
// ---------------------------------------------------------------------------
__global__ __launch_bounds__(256) void prep_shapelets(const float* __restrict__ sh,
                                                      _Float16* __restrict__ spack)
{
    __shared__ _Float16 snorm[NSH * L_LEN];   // 32 KB
    const int tid = threadIdx.x;
    if (tid < NSH) {
        const float* row = sh + tid * L_LEN;
        float s1 = 0.f, s2 = 0.f;
        for (int l = 0; l < L_LEN; ++l) { float v = row[l]; s1 += v; s2 += v * v; }
        const float mu  = s1 * (1.0f / L_LEN);
        const float var = fmaxf(s2 * (1.0f / L_LEN) - mu * mu, 0.f);
        const float inv = 1.0f / (sqrtf(var) + EPS_F);
        for (int l = 0; l < L_LEN; ++l)
            snorm[tid * L_LEN + l] = (_Float16)((row[l] - mu) * inv);
    }
    __syncthreads();
    #pragma unroll
    for (int it = 0; it < 4; ++it) {
        const int e    = tid + it * 256;        // 0..1023 operand-lane entries
        const int lane = e & 31;
        const int kc   = (e >> 5) & 3;
        const int nt   = e >> 7;
        const int n    = nt * 16 + (lane & 15);
        const int kb   = kc * 32 + ((lane >> 4) << 3);
        V16HU u;
        #pragma unroll
        for (int j = 0; j < 16; ++j)
            u.v[j] = snorm[n * L_LEN + kb + j + ((j >> 3) << 3)];
        ((v16h*)spack)[e] = u.v;
    }
}

// ---------------------------------------------------------------------------
// Kernel 2: one block per batch row; 8 waves; wave w owns shapelets [16w,16w+16).
// ---------------------------------------------------------------------------
__global__ __launch_bounds__(256) void shapelet_corr(const float* __restrict__ x,
                                                     const _Float16* __restrict__ spack,
                                                     float* __restrict__ out)
{
    __shared__ __align__(16) _Float16 xrep[REPS * RSTRIDE];  // 32896 B
    __shared__ float sbuf[NBLK * 16];                        // 15936 B (scale per window)

    const int tid  = threadIdx.x;
    const int b    = blockIdx.x;
    const int lane = tid & 31;
    const int wid  = tid >> 5;
    const float* xg = x + b * T_LEN;

    // --- stage row into f16, 4 shifted replicas: xrep[r][j] = x[j + r] -------
    {
        const float4* xg4 = (const float4*)xg;
        float vals[16];
        #pragma unroll
        for (int q4 = 0; q4 < 4; ++q4) {
            float4 v = xg4[tid * 4 + q4];
            vals[q4 * 4 + 0] = v.x; vals[q4 * 4 + 1] = v.y;
            vals[q4 * 4 + 2] = v.z; vals[q4 * 4 + 3] = v.w;
        }
        const int e0 = tid * 16;
        #pragma unroll
        for (int q = 0; q < 16; ++q) {
            const _Float16 hv = (_Float16)vals[q];
            const int e = e0 + q;
            #pragma unroll
            for (int r = 0; r < REPS; ++r) {
                const int j = e - r;
                if (j >= 0) xrep[r * RSTRIDE + j] = hv;
            }
        }
        if (tid < 128) {                         // zero the tails past x[4095-r]
            const int r = tid >> 5;
            const int j = 4080 + (tid & 31);
            if (j >= T_LEN - r) xrep[r * RSTRIDE + j] = (_Float16)0.f;
        }
    }

    // --- per-window scale = 1/(128*(sqrt(var)+eps)) via sliding sums (f32) ---
    if (tid < NBLK) {
        const int w0 = tid * 16;
        float s1 = 0.f, s2 = 0.f;
        for (int e = 0; e < L_LEN; ++e) { float v = xg[w0 + e]; s1 += v; s2 += v * v; }
        #pragma unroll 1
        for (int m = 0; m < 16; ++m) {
            const int t = w0 + m;
            float sc = 0.f;
            if (t < NWIN) {
                const float mu  = s1 * (1.0f / L_LEN);
                const float var = fmaxf(s2 * (1.0f / L_LEN) - mu * mu, 0.f);
                sc = 1.0f / ((sqrtf(var) + EPS_F) * (float)L_LEN);
            }
            sbuf[t] = sc;
            const int   ia = t + L_LEN;
            const float va = (ia < T_LEN) ? xg[ia] : 0.f;
            const float vr = xg[t];
            s1 += va - vr;
            s2 += va * va - vr * vr;
        }
    }
    __syncthreads();

    // --- B operands resident in registers (4 K-chunks of 32) -----------------
    v16h bm[4];
    #pragma unroll
    for (int kc = 0; kc < 4; ++kc)
        bm[kc] = ((const v16h*)spack)[(wid * 4 + kc) * 32 + lane];

    // A fragment base: lane l: M=l&15, h=l>>4; group start (halfs) is
    // t0 + M + kc*32 + 8h; replica r = M&3 makes it 8B aligned.
    const int h     = lane >> 4;
    const int abase = (lane & 3) * RSTRIDE + 4 * ((lane >> 2) & 3) + 8 * h;

    float rmax[8];
    #pragma unroll
    for (int i = 0; i < 8; ++i) rmax[i] = NEG_BIG;

    for (int blk = 0; blk < NBLK; ++blk) {
        const int t0 = blk * 16;
        v8f c = {};
        #pragma unroll
        for (int kc = 0; kc < 4; ++kc) {
            const int p = abase + t0 + kc * 32;
            const uint2 g0 = *(const uint2*)&xrep[p];        // K = 8h + 0..3
            const uint2 g1 = *(const uint2*)&xrep[p + 4];    // K = 8h + 4..7
            const uint2 g2 = *(const uint2*)&xrep[p + 16];   // K = 16+8h+0..3
            const uint2 g3 = *(const uint2*)&xrep[p + 20];   // K = 16+8h+4..7
            V16HU a;
            a.u4[0] = make_uint4(g0.x, g0.y, g1.x, g1.y);
            a.u4[1] = make_uint4(g2.x, g2.y, g3.x, g3.y);
            c = __builtin_amdgcn_wmma_f32_16x16x32_f16(
                    false, a.v, false, bm[kc], (short)0, c, false, false);
        }
        // scale by inv-std, mask invalid windows, running max
        #pragma unroll
        for (int i = 0; i < 8; ++i) {
            const int t = t0 + i + 8 * h;             // C layout: VGPR i -> M=i / 8+i
            float v = c[i] * sbuf[t];
            v = (t < NWIN) ? v : NEG_BIG;
            rmax[i] = fmaxf(rmax[i], v);
        }
    }

    // reduce 8 M-rows, then swap lane halves (same N, different M) and store
    float m = rmax[0];
    #pragma unroll
    for (int i = 1; i < 8; ++i) m = fmaxf(m, rmax[i]);
    const float other =
        __int_as_float(__builtin_amdgcn_ds_swizzle(__float_as_int(m), 0x401f)); // SWAPX16
    m = fmaxf(m, other);
    if (lane < 16)
        out[b * NSH + wid * 16 + (lane & 15)] = m;
}

// ---------------------------------------------------------------------------
extern "C" void kernel_launch(void* const* d_in, const int* in_sizes, int n_in,
                              void* d_out, int out_size, void* d_ws, size_t ws_size,
                              hipStream_t stream)
{
    (void)in_sizes; (void)n_in; (void)out_size; (void)ws_size;
    const float* x  = (const float*)d_in[0];   // (256, 4096) f32
    const float* sh = (const float*)d_in[1];   // (8, 16, 128) f32
    float* out      = (float*)d_out;           // (256, 128) f32
    _Float16* spack = (_Float16*)d_ws;         // 32 KB packed B operands

    prep_shapelets<<<1, 256, 0, stream>>>(sh, spack);
    shapelet_corr<<<256, 256, 0, stream>>>(x, spack, out);
}
